// LongContextAttention_11424613007886
// MI455X (gfx1250) — matrix operand order repaired and checked by reading, not compiled, not run on hardware
//
#include <hip/hip_runtime.h>
#include <hip/hip_bf16.h>

typedef __attribute__((ext_vector_type(16))) _Float16 v16h;
typedef __attribute__((ext_vector_type(8)))  _Float16 v8h;
typedef __attribute__((ext_vector_type(4)))  _Float16 v4h;
typedef __attribute__((ext_vector_type(8)))  float    v8f;
typedef __attribute__((ext_vector_type(4)))  int      v4i;

#if defined(__has_builtin)
# if __has_builtin(__builtin_amdgcn_global_load_async_to_lds_b128)
#  define HAVE_ASYNC_COPY 1
# endif
#endif
#ifndef HAVE_ASYNC_COPY
# define HAVE_ASYNC_COPY 0
#endif

#if defined(__has_builtin)
# if __has_builtin(__builtin_amdgcn_s_wait_asynccnt)
#  define ASYNC_WAIT(n) __builtin_amdgcn_s_wait_asynccnt(n)
# endif
#endif
#ifndef ASYNC_WAIT
# define ASYNC_WAIT(n) asm volatile("s_wait_asynccnt " #n ::: "memory")
#endif

// exact pointer types expected by the async-copy builtin (from clang diagnostic):
//   arg0: addrspace(1) v4i*   (global source)
//   arg1: addrspace(3) v4i*   (LDS destination)
#define GV4IP(p) ((__attribute__((address_space(1))) v4i*)(p))
#define LV4IP(p) ((__attribute__((address_space(3))) v4i*)(p))

namespace {
constexpr int S_ = 4096;
constexpr int H_ = 16;
constexpr int D_ = 128;
constexpr int BM = 64;    // query rows per workgroup (16 per wave)
constexpr int BN = 32;    // key rows per iteration
// softmax scale (1/sqrt(128)) folded with log2(e) so we can use exp2
constexpr float kScale = 0.0883883476483184405f * 1.44269504088896340f;

__device__ __forceinline__ v16h ld16(const _Float16* p) {
  v8h lo = *(const v8h*)p;
  v8h hi = *(const v8h*)(p + 8);
  v16h r;
#pragma unroll
  for (int i = 0; i < 8; ++i) { r[i] = lo[i]; r[i + 8] = hi[i]; }
  return r;
}

__device__ __forceinline__ v16h ld16_split(const _Float16* p0, const _Float16* p1) {
  v8h lo = *(const v8h*)p0;
  v8h hi = *(const v8h*)p1;
  v16h r;
#pragma unroll
  for (int i = 0; i < 8; ++i) { r[i] = lo[i]; r[i + 8] = hi[i]; }
  return r;
}

__device__ __forceinline__ void cvt8(v16h& d, int base, float4 a, float4 b) {
  d[base + 0] = (_Float16)a.x; d[base + 1] = (_Float16)a.y;
  d[base + 2] = (_Float16)a.z; d[base + 3] = (_Float16)a.w;
  d[base + 4] = (_Float16)b.x; d[base + 5] = (_Float16)b.y;
  d[base + 6] = (_Float16)b.z; d[base + 7] = (_Float16)b.w;
}
} // namespace

// ---------------- pre-pass: K f32 [S,H,D] -> f16 [H][S][D] ----------------
extern "C" __global__ __launch_bounds__(256)
void cvt_k_kernel(const float* __restrict__ K, _Float16* __restrict__ KH) {
  const int c  = blockIdx.x * 256 + threadIdx.x;   // 2^20 threads total
  const int d8 = (c & 15) * 8;
  const int s  = (c >> 4) & (S_ - 1);
  const int h  = c >> 16;
  const float* src = K + ((size_t)s * H_ + h) * D_ + d8;
  float4 f0 = ((const float4*)src)[0];
  float4 f1 = ((const float4*)src)[1];
  v8h o;
  o[0] = (_Float16)f0.x; o[1] = (_Float16)f0.y;
  o[2] = (_Float16)f0.z; o[3] = (_Float16)f0.w;
  o[4] = (_Float16)f1.x; o[5] = (_Float16)f1.y;
  o[6] = (_Float16)f1.z; o[7] = (_Float16)f1.w;
  *(v8h*)(KH + ((size_t)h * S_ + s) * D_ + d8) = o;
}

// -------- pre-pass: V f32 [S,H,D] -> f16 transposed [H][D][S] (LDS tile) ----
extern "C" __global__ __launch_bounds__(256)
void cvt_vt_kernel(const float* __restrict__ V, _Float16* __restrict__ VT) {
  __shared__ alignas(16) _Float16 T[64][72];   // pitch 72 halves
  const int s0 = blockIdx.x * 64;
  const int d0 = blockIdx.y * 64;
  const int h  = blockIdx.z;
  const int t  = threadIdx.x;
  {
    const int srow = t >> 2;
    const int dseg = (t & 3) * 16;
    const float* src = V + ((size_t)(s0 + srow) * H_ + h) * D_ + d0 + dseg;
#pragma unroll
    for (int i = 0; i < 4; ++i) {
      float4 f = ((const float4*)src)[i];
      v4h q;
      q[0] = (_Float16)f.x; q[1] = (_Float16)f.y;
      q[2] = (_Float16)f.z; q[3] = (_Float16)f.w;
      *(v4h*)&T[srow][dseg + 4 * i] = q;
    }
  }
  __syncthreads();
  {
    const int drow = t >> 2;
    const int sseg = (t & 3) * 16;
    _Float16* dst = VT + ((size_t)h * D_ + d0 + drow) * S_ + s0 + sseg;
    v8h a, b;
#pragma unroll
    for (int i = 0; i < 8; ++i) {
      a[i] = T[sseg + i][drow];
      b[i] = T[sseg + 8 + i][drow];
    }
    *(v8h*)dst = a;
    *(v8h*)(dst + 8) = b;
  }
}

// --------------------------- flash attention ------------------------------
template <bool USE_WS>
__global__ __launch_bounds__(128)
void fa_fwd_causal(const float* __restrict__ Q,
                   const float* __restrict__ K,
                   const float* __restrict__ V,
                   const _Float16* __restrict__ KH,   // f16 [H][S][D]
                   const _Float16* __restrict__ VT,   // f16 [H][D][S]
                   float* __restrict__ O) {
  __shared__ alignas(16) _Float16 Ksh[2][BN * D_];    // [n][d], 2 x 8KB
  __shared__ alignas(16) _Float16 Vtsh[2][D_ * BN];   // [d][n], 2 x 8KB
  __shared__ alignas(16) _Float16 Psh[4][16 * BN];    // per-wave P tile, 4KB

  const int tid      = threadIdx.x;
  const int wave     = tid >> 5;
  const int lane     = tid & 31;
  const int lane16   = lane & 15;
  const int laneHalf = lane >> 4;
  const int h        = blockIdx.y;
  const int q0       = blockIdx.x * BM;
  const int wrow0    = q0 + wave * 16;

  // ---- Q fragments (A-matrix layout, f32 -> f16 in registers, loaded once) ----
  v16h aq[4];
  {
    const float* qrow = Q + ((size_t)(wrow0 + lane16) * H_ + h) * D_;
#pragma unroll
    for (int c = 0; c < 4; ++c) {
      const int d0 = 32 * c + laneHalf * 8;
      float4 a0 = *(const float4*)(qrow + d0);
      float4 a1 = *(const float4*)(qrow + d0 + 4);
      float4 b0 = *(const float4*)(qrow + d0 + 16);
      float4 b1 = *(const float4*)(qrow + d0 + 20);
      cvt8(aq[c], 0, a0, a1);
      cvt8(aq[c], 8, b0, b1);
    }
  }

  v8f acc[8];
  float mrow[8], lrow[8];
#pragma unroll
  for (int dt = 0; dt < 8; ++dt)
#pragma unroll
    for (int r = 0; r < 8; ++r) acc[dt][r] = 0.0f;
#pragma unroll
  for (int r = 0; r < 8; ++r) { mrow[r] = -3.0e38f; lrow[r] = 0.0f; }

  const int nkb = (q0 + BM) / BN;   // causal clip: keys 0 .. q0+63  (>= 2)

  // ---- staging: fill LDS buffer `buf` with key block jj ----
  auto issue_copy = [&](int jj, int buf) {
    const int kb = jj * BN;
    if constexpr (USE_WS) {
      const char* kg = (const char*)(KH + ((size_t)h * S_ + kb) * D_);  // 8KB contig
      char* kd = (char*)&Ksh[buf][0];
      const _Float16* vg = VT + (size_t)h * D_ * S_;
      _Float16* vd = &Vtsh[buf][0];
#if HAVE_ASYNC_COPY
#pragma unroll
      for (int i = 0; i < 4; ++i) {
        const int id = i * 128 + tid;
        __builtin_amdgcn_global_load_async_to_lds_b128(
            GV4IP(kg + id * 16), LV4IP(kd + id * 16), 0, 0);
      }
#pragma unroll
      for (int i = 0; i < 4; ++i) {
        const int id  = i * 128 + tid;
        const int row = id >> 2;          // d row 0..127
        const int c8  = (id & 3) * 8;     // 8-half chunk within 32
        __builtin_amdgcn_global_load_async_to_lds_b128(
            GV4IP(vg + ((size_t)row * S_ + kb + c8)),
            LV4IP(vd + row * BN + c8), 0, 0);
      }
#else
#pragma unroll
      for (int i = 0; i < 4; ++i) {
        const int id = i * 128 + tid;
        *(v8h*)(kd + id * 16) = *(const v8h*)(kg + id * 16);
      }
#pragma unroll
      for (int i = 0; i < 4; ++i) {
        const int id  = i * 128 + tid;
        const int row = id >> 2;
        const int c8  = (id & 3) * 8;
        *(v8h*)(vd + row * BN + c8) = *(const v8h*)(vg + (size_t)row * S_ + kb + c8);
      }
#endif
    } else {
      // self-contained path: f32 global -> f16 LDS, V transposed on the fly
      const int row   = tid >> 2;
      const int dbase = (tid & 3) * 32;
      const float* kp = K + ((size_t)(kb + row) * H_ + h) * D_ + dbase;
      const float* vp = V + ((size_t)(kb + row) * H_ + h) * D_ + dbase;
#pragma unroll
      for (int g = 0; g < 4; ++g) {
        float4 f0 = ((const float4*)kp)[2 * g];
        float4 f1 = ((const float4*)kp)[2 * g + 1];
        v8h h8;
        h8[0] = (_Float16)f0.x; h8[1] = (_Float16)f0.y;
        h8[2] = (_Float16)f0.z; h8[3] = (_Float16)f0.w;
        h8[4] = (_Float16)f1.x; h8[5] = (_Float16)f1.y;
        h8[6] = (_Float16)f1.z; h8[7] = (_Float16)f1.w;
        *(v8h*)(&Ksh[buf][0] + row * D_ + dbase + 8 * g) = h8;
      }
#pragma unroll
      for (int i = 0; i < 8; ++i) {
        float4 f = ((const float4*)vp)[i];
        const int d0 = dbase + 4 * i;
        _Float16* vd = &Vtsh[buf][0];
        vd[(d0 + 0) * BN + row] = (_Float16)f.x;
        vd[(d0 + 1) * BN + row] = (_Float16)f.y;
        vd[(d0 + 2) * BN + row] = (_Float16)f.z;
        vd[(d0 + 3) * BN + row] = (_Float16)f.w;
      }
    }
  };

  issue_copy(0, 0);   // prologue

  for (int j = 0; j < nkb; ++j) {
    const int cur = j & 1;
    const bool have_next = (j + 1 < nkb);
    if (have_next) issue_copy(j + 1, cur ^ 1);

    if constexpr (USE_WS) {
#if HAVE_ASYNC_COPY
      if (have_next) { ASYNC_WAIT(8); } else { ASYNC_WAIT(0); }
#endif
    }
    __syncthreads();   // block j fully in LDS for all waves

    const int kb = j * BN;
    const bool active = (kb <= wrow0 + 15);   // wave-uniform causal early-out
    if (active) {
      // ---- scores = Q . K^T  (8x v_wmma_f32_16x16x32_f16) ----
      v8f sc[2];
#pragma unroll
      for (int nt = 0; nt < 2; ++nt) {
        v8f c;
#pragma unroll
        for (int r = 0; r < 8; ++r) c[r] = 0.0f;
#pragma unroll
        for (int cc = 0; cc < 4; ++cc) {
          const _Float16* bp =
              &Ksh[cur][0] + (nt * 16 + lane16) * D_ + cc * 32 + laneHalf * 8;
          v16h bk = ld16_split(bp, bp + 16);
          c = __builtin_amdgcn_wmma_f32_16x16x32_f16(
              false, aq[cc], false, bk, (short)0, c, false, false);
        }
        sc[nt] = c;
      }

      // ---- causal mask ----
      if (kb + BN - 1 > wrow0) {
#pragma unroll
        for (int nt = 0; nt < 2; ++nt)
#pragma unroll
          for (int r = 0; r < 8; ++r) {
            const int rowg = wrow0 + laneHalf * 8 + r;
            const int colg = kb + nt * 16 + lane16;
            if (colg > rowg) sc[nt][r] = -3.0e38f;
          }
      }

      // ---- row max across 16-lane group ----
      float pm[8], ps[8];
#pragma unroll
      for (int r = 0; r < 8; ++r) {
        float m = fmaxf(sc[0][r], sc[1][r]);
        m = fmaxf(m, __shfl_xor(m, 1, 32));
        m = fmaxf(m, __shfl_xor(m, 2, 32));
        m = fmaxf(m, __shfl_xor(m, 4, 32));
        m = fmaxf(m, __shfl_xor(m, 8, 32));
        pm[r] = m;
      }

      // ---- online softmax update ----
#pragma unroll
      for (int r = 0; r < 8; ++r) {
        const float mn    = fmaxf(mrow[r], pm[r]);
        const float alpha = __builtin_amdgcn_exp2f((mrow[r] - mn) * kScale);
        mrow[r] = mn;
        const float p0 = __builtin_amdgcn_exp2f((sc[0][r] - mn) * kScale);
        const float p1 = __builtin_amdgcn_exp2f((sc[1][r] - mn) * kScale);
        sc[0][r] = p0; sc[1][r] = p1;
        ps[r] = p0 + p1;
#pragma unroll
        for (int dt = 0; dt < 8; ++dt) acc[dt][r] *= alpha;
        lrow[r] *= alpha;
      }
#pragma unroll
      for (int r = 0; r < 8; ++r) {
        float s = ps[r];
        s += __shfl_xor(s, 1, 32);
        s += __shfl_xor(s, 2, 32);
        s += __shfl_xor(s, 4, 32);
        s += __shfl_xor(s, 8, 32);
        lrow[r] += s;
      }

      // ---- stage P to LDS (C-layout -> A-layout), f32 -> f16 ----
      _Float16* pw = &Psh[wave][0];
#pragma unroll
      for (int nt = 0; nt < 2; ++nt)
#pragma unroll
        for (int r = 0; r < 8; ++r)
          pw[(laneHalf * 8 + r) * BN + nt * 16 + lane16] = (_Float16)sc[nt][r];
      asm volatile("s_wait_dscnt 0" ::: "memory");

      const _Float16* pr = pw + lane16 * BN + laneHalf * 8;
      v16h ap = ld16_split(pr, pr + 16);

      // ---- out += P . V  (8x v_wmma_f32_16x16x32_f16) ----
#pragma unroll
      for (int dt = 0; dt < 8; ++dt) {
        const _Float16* vb = &Vtsh[cur][0] + (dt * 16 + lane16) * BN + laneHalf * 8;
        v16h bv = ld16_split(vb, vb + 16);
        acc[dt] = __builtin_amdgcn_wmma_f32_16x16x32_f16(
            false, ap, false, bv, (short)0, acc[dt], false, false);
      }
    }
    __syncthreads();   // everyone done reading buf `cur` before it is refilled
  }

  // ---- normalize and store (fp32 output, [S,H,D]) ----
  float inv[8];
#pragma unroll
  for (int r = 0; r < 8; ++r) inv[r] = __builtin_amdgcn_rcpf(lrow[r]);
#pragma unroll
  for (int dt = 0; dt < 8; ++dt)
#pragma unroll
    for (int r = 0; r < 8; ++r) {
      const int rowg = wrow0 + laneHalf * 8 + r;
      const int colg = dt * 16 + lane16;
      O[((size_t)rowg * H_ + h) * D_ + colg] = acc[dt][r] * inv[r];
    }
}

extern "C" void kernel_launch(void* const* d_in, const int* in_sizes, int n_in,
                              void* d_out, int out_size, void* d_ws, size_t ws_size,
                              hipStream_t stream) {
  (void)in_sizes; (void)n_in; (void)out_size;
  const float* Q = (const float*)d_in[0];
  const float* K = (const float*)d_in[1];
  const float* V = (const float*)d_in[2];
  float* O = (float*)d_out;

  const size_t elems = (size_t)H_ * S_ * D_;
  const size_t need  = elems * sizeof(_Float16) * 2;   // f16 K + f16 V^T

  dim3 grid(S_ / BM, H_, 1);
  dim3 block(128, 1, 1);

  if (ws_size >= need) {
    _Float16* KH = (_Float16*)d_ws;
    _Float16* VT = KH + elems;
    cvt_k_kernel<<<dim3((unsigned)(elems / 8 / 256)), dim3(256), 0, stream>>>(K, KH);
    cvt_vt_kernel<<<dim3(S_ / 64, D_ / 64, H_), dim3(256), 0, stream>>>(V, VT);
    fa_fwd_causal<true><<<grid, block, 0, stream>>>(Q, K, V, KH, VT, O);
  } else {
    fa_fwd_causal<false><<<grid, block, 0, stream>>>(Q, K, V, nullptr, nullptr, O);
  }
}
